// RVMB_20040317403815
// MI455X (gfx1250) — matrix operand order repaired and checked by reading, not compiled, not run on hardware
//
#include <hip/hip_runtime.h>

// ---------------- problem constants ----------------
#define BB   2
#define Lc   16384           // F*H*W tokens per batch
#define NTc  32768           // B*L total tokens
#define Cc   64
#define DIc  96
#define DSc  16
#define DRc  4
#define DCc  4
#define MHc  96
#define N1c  192             // 2*DI
#define CSc  128             // scan chunk size
#define NCHc 128             // chunks per batch (L/CS)

typedef __attribute__((ext_vector_type(16))) __bf16 v16bf;
typedef __attribute__((ext_vector_type(8)))  __bf16 v8bf;
typedef __attribute__((ext_vector_type(4)))  __bf16 v4bf;
typedef __attribute__((ext_vector_type(8)))  float  v8f;

static __device__ inline v8f wmma_bf16(v16bf a, v16bf b, v8f c) {
  return __builtin_amdgcn_wmma_f32_16x16x32_bf16(false, a, false, b, (short)0, c, false, false);
}

// Contiguous per-lane fragment load: 32 bytes -> 2x b128 loads.
static __device__ inline v16bf frag_ld(const __bf16* p) {
  v8bf lo = *(const v8bf*)p;
  v8bf hi = *(const v8bf*)(p + 8);
  return __builtin_shufflevector(lo, hi, 0,1,2,3,4,5,6,7,8,9,10,11,12,13,14,15);
}

// Swizzled A-side element index for (row r, col c) with KT k-tiles per row band.
// A frag (wave32): lanes 0-15: M=lane, K in {0..7}U{16..23}; lanes 16-31: K in {8..15}U{24..31}
static __device__ inline int aswz_idx(int r, int c, int KT) {
  int kl = c & 31;
  int hi = (kl >> 3) & 1;
  int j  = (kl & 7) | ((kl >> 4) << 3);
  return ((((r >> 4) * KT + (c >> 5)) * 32) + ((r & 15) | (hi << 4))) * 16 + j;
}

static __device__ inline float sigmoidf_(float v) { return 1.f / (1.f + __expf(-v)); }

// Swizzle a KxNsrc f32 weight matrix into B-fragment bf16 layout [kt][nt][lane][16] (N padded to Npad).
static __device__ void swz_weight(const float* __restrict__ W, int K, int Nsrc, int Npad,
                                  __bf16* __restrict__ out, int gid, int np) {
  int NT = Npad >> 4;
  int total = K * Npad;
  for (int i = gid; i < total; i += np) {
    int j    = i & 15;
    int lane = (i >> 4) & 31;
    int t    = i >> 9;                // kt*NT + nt
    int nt = t % NT, kt = t / NT;
    int k = (kt << 5) + ((j >> 3) << 4) + ((lane >> 4) << 3) + (j & 7);   // lanes 0-15: K in {0..7}U{16..23}; 16-31: {8..15}U{24..31}
    int n = (nt << 4) + (lane & 15);
    out[i] = (__bf16)((n < Nsrc) ? W[k * Nsrc + n] : 0.f);
  }
}

// ---------------- K0: fold modulation, swizzle weights to bf16 fragments ----------------
__global__ void k0_prep(const float* __restrict__ cdp, const float* __restrict__ vk1,
                        const float* __restrict__ vk2,
                        const float* __restrict__ ln1w, const float* __restrict__ ln1b,
                        const float* __restrict__ ln2w, const float* __restrict__ ln2b,
                        const float* __restrict__ in_proj_w, const float* __restrict__ x_proj_w,
                        const float* __restrict__ out_proj_w, const float* __restrict__ fc1w,
                        const float* __restrict__ fc2w, const float* __restrict__ A_log,
                        float* __restrict__ w1e, float* __restrict__ b1e,
                        float* __restrict__ w2e, float* __restrict__ b2e,
                        float* __restrict__ Amat,
                        __bf16* __restrict__ inprojb, __bf16* __restrict__ xprojb,
                        __bf16* __restrict__ outprojb, __bf16* __restrict__ fc1bw,
                        __bf16* __restrict__ fc2bw) {
  int gid = blockIdx.x * blockDim.x + threadIdx.x;
  int np  = gridDim.x * blockDim.x;
  // ((xn)*w + b)*c1 + c2 = xn*(w*c1) + (b*c1 + c2)
  for (int i = gid; i < BB * Cc; i += np) {
    int b = i / Cc, c = i - b * Cc;
    float m1 = 0.f, m1s = 0.f, m2 = 0.f, m2s = 0.f;
    for (int k = 0; k < 128; ++k) {
      float cv = cdp[b * 128 + k];
      m1  += cv * vk1[k * 128 + c];
      m1s += cv * vk1[k * 128 + 64 + c];
      m2  += cv * vk2[k * 128 + c];
      m2s += cv * vk2[k * 128 + 64 + c];
    }
    w1e[i] = ln1w[c] * m1;  b1e[i] = ln1b[c] * m1 + m1s;
    w2e[i] = ln2w[c] * m2;  b2e[i] = ln2b[c] * m2 + m2s;
  }
  for (int i = gid; i < DIc * DSc; i += np) Amat[i] = -__expf(A_log[i]);
  swz_weight(in_proj_w,  Cc,  N1c, N1c, inprojb,  gid, np);   // 64x192, KT=2 NT=12
  swz_weight(x_proj_w,   DIc, 36,  48,  xprojb,   gid, np);   // 96x36 -> 96x48, KT=3 NT=3
  swz_weight(out_proj_w, DIc, Cc,  Cc,  outprojb, gid, np);   // 96x64, KT=3 NT=4
  swz_weight(fc1w,       Cc,  MHc, MHc, fc1bw,    gid, np);   // 64x96, KT=2 NT=6
  swz_weight(fc2w,       MHc, Cc,  Cc,  fc2bw,    gid, np);   // 96x64, KT=3 NT=4
}

// ---------------- K1: LN1 + modulation + in_proj (64 -> 192) ----------------
__global__ void k1_ln_inproj(const float* __restrict__ x, const float* __restrict__ w1e,
                             const float* __restrict__ b1e, const __bf16* __restrict__ wswz,
                             float* __restrict__ xs_pre, float* __restrict__ zout) {
  extern __shared__ char smem[];
  float*  xsh = (float*)smem;                       // 128 x 68 f32
  __bf16* As  = (__bf16*)(smem + 128 * 68 * 4);     // 128 x 64 bf16 (frag-swizzled, KT=2)
  float*  mu  = (float*)((char*)As + 128 * 64 * 2); // 128
  float*  rs  = mu + 128;                           // 128
  int tid = threadIdx.x;
  int t0  = blockIdx.x * 128;
  int bat = t0 / Lc;
  const float4* xg = (const float4*)x;
  float4* xsh4 = (float4*)xsh;                      // stride 17 float4
  for (int i = tid; i < 128 * 16; i += 256) {
    int r = i >> 4, c = i & 15;
    xsh4[r * 17 + c] = xg[(size_t)(t0 + r) * 16 + c];
  }
  __syncthreads();
  if (tid < 128) {
    float s = 0.f, ss = 0.f;
    const float4* p = xsh4 + tid * 17;
#pragma unroll
    for (int c = 0; c < 16; ++c) {
      float4 v = p[c];
      s  += v.x + v.y + v.z + v.w;
      ss += v.x * v.x + v.y * v.y + v.z * v.z + v.w * v.w;
    }
    float m = s * (1.f / Cc);
    mu[tid] = m;
    rs[tid] = rsqrtf(ss * (1.f / Cc) - m * m + 1e-6f);
  }
  __syncthreads();
  for (int i = tid; i < 128 * 16; i += 256) {
    int r = i >> 4, c4 = i & 15, c0 = c4 * 4;
    float4 v = xsh4[r * 17 + c4];
    float m = mu[r], rr = rs[r];
    const float* we = w1e + bat * Cc + c0;
    const float* be = b1e + bat * Cc + c0;
    v4bf o;
    o[0] = (__bf16)((v.x - m) * rr * we[0] + be[0]);
    o[1] = (__bf16)((v.y - m) * rr * we[1] + be[1]);
    o[2] = (__bf16)((v.z - m) * rr * we[2] + be[2]);
    o[3] = (__bf16)((v.w - m) * rr * we[3] + be[3]);
    *(v4bf*)(As + aswz_idx(r, c0, 2)) = o;
  }
  __syncthreads();
  int wave = tid >> 5, lane = tid & 31;
  v8f acc[12] = {};
#pragma unroll
  for (int kt = 0; kt < 2; ++kt) {
    v16bf a = frag_ld(As + ((wave * 2 + kt) * 32 + lane) * 16);
#pragma unroll
    for (int nt = 0; nt < 12; ++nt)
      acc[nt] = wmma_bf16(a, frag_ld(wswz + ((kt * 12 + nt) * 32 + lane) * 16), acc[nt]);
  }
#pragma unroll
  for (int nt = 0; nt < 12; ++nt) {
    int col = nt * 16 + (lane & 15);
    int rb  = t0 + wave * 16 + ((lane >> 4) << 3);
    float* dst; int cc;
    if (col < DIc) { dst = xs_pre; cc = col; } else { dst = zout; cc = col - DIc; }
#pragma unroll
    for (int v = 0; v < 8; ++v) dst[(size_t)(rb + v) * DIc + cc] = acc[nt][v];
  }
}

// ---------------- K2: causal depthwise conv + silu + x_proj + dt_proj ----------------
__global__ void k2_conv_xproj(const float* __restrict__ xs_pre, const float* __restrict__ conv_w,
                              const float* __restrict__ conv_b, const __bf16* __restrict__ xprojswz,
                              const float* __restrict__ dtw, const float* __restrict__ dtb,
                              float* __restrict__ xs, float* __restrict__ dt,
                              float* __restrict__ Bm, float* __restrict__ Cm) {
  extern __shared__ char smem[];
  float*  xp   = (float*)smem;                        // 131 x 100 f32
  __bf16* As   = (__bf16*)(smem + 131 * 100 * 4);     // 128 x 96 bf16 (frag-swizzled, KT=3)
  float*  dbcs = (float*)((char*)As + 128 * 96 * 2);  // 128 x 48 f32
  int tid = threadIdx.x;
  int t0  = blockIdx.x * 128;
  int p0  = t0 & (Lc - 1);
  float4* xp4 = (float4*)xp;                          // stride 25 float4
  const float4* xpre4 = (const float4*)xs_pre;
  for (int i = tid; i < 131 * 24; i += 256) {
    int r = i / 24, c = i - r * 24;
    int p = p0 - 3 + r;                               // causal left zero-pad at batch start
    float4 v = {0.f, 0.f, 0.f, 0.f};
    if (p >= 0) v = xpre4[(size_t)(t0 - 3 + r) * 24 + c];
    xp4[r * 25 + c] = v;
  }
  __syncthreads();
  for (int i = tid; i < 128 * 24; i += 256) {
    int r = i / 24, q = i - r * 24, d0 = q * 4;
    float4 rowv[4];
#pragma unroll
    for (int j = 0; j < DCc; ++j) rowv[j] = xp4[(r + j) * 25 + q];
    const float* rv = (const float*)rowv;             // rv[j*4+cc]
    float res[4];
#pragma unroll
    for (int ccm = 0; ccm < 4; ++ccm) {
      int d = d0 + ccm;
      float v = conv_b[d];
#pragma unroll
      for (int j = 0; j < DCc; ++j) v += conv_w[d * DCc + j] * rv[j * 4 + ccm];
      res[ccm] = v * sigmoidf_(v);                    // silu
    }
    float4 f4; f4.x = res[0]; f4.y = res[1]; f4.z = res[2]; f4.w = res[3];
    *(float4*)&xs[(size_t)(t0 + r) * DIc + d0] = f4;
    v4bf o;
    o[0] = (__bf16)res[0]; o[1] = (__bf16)res[1]; o[2] = (__bf16)res[2]; o[3] = (__bf16)res[3];
    *(v4bf*)(As + aswz_idx(r, d0, 3)) = o;
  }
  __syncthreads();
  int wave = tid >> 5, lane = tid & 31;
  v8f acc[3] = {};
#pragma unroll
  for (int kt = 0; kt < 3; ++kt) {
    v16bf a = frag_ld(As + ((wave * 3 + kt) * 32 + lane) * 16);
#pragma unroll
    for (int nt = 0; nt < 3; ++nt)
      acc[nt] = wmma_bf16(a, frag_ld(xprojswz + ((kt * 3 + nt) * 32 + lane) * 16), acc[nt]);
  }
#pragma unroll
  for (int nt = 0; nt < 3; ++nt) {
    int col = nt * 16 + (lane & 15);
    int rb  = wave * 16 + ((lane >> 4) << 3);
#pragma unroll
    for (int v = 0; v < 8; ++v) dbcs[(rb + v) * 48 + col] = acc[nt][v];
  }
  __syncthreads();
  // dt = softplus(dbc[:, :4] @ dt_proj_w + dt_proj_b)
  for (int i = tid; i < 128 * 24; i += 256) {
    int r = i / 24, q = i - r * 24, d0 = q * 4;
    float4 dv = *(const float4*)&dbcs[r * 48];
    float res[4];
#pragma unroll
    for (int ccm = 0; ccm < 4; ++ccm) {
      int d = d0 + ccm;
      float v = dtb[d] + dv.x * dtw[0 * DIc + d] + dv.y * dtw[1 * DIc + d]
                       + dv.z * dtw[2 * DIc + d] + dv.w * dtw[3 * DIc + d];
      res[ccm] = (v > 20.f) ? v : log1pf(__expf(v));
    }
    float4 f4; f4.x = res[0]; f4.y = res[1]; f4.z = res[2]; f4.w = res[3];
    *(float4*)&dt[(size_t)(t0 + r) * DIc + d0] = f4;
  }
  for (int i = tid; i < 128 * 4; i += 256) {
    int r = i >> 2, s4 = (i & 3) * 4;
    *(float4*)&Bm[(size_t)(t0 + r) * DSc + s4] = *(const float4*)&dbcs[r * 48 + 4 + s4];
    *(float4*)&Cm[(size_t)(t0 + r) * DSc + s4] = *(const float4*)&dbcs[r * 48 + 20 + s4];
  }
}

// ---------------- K3a: per-chunk local scan aggregates ----------------
__global__ void k3a_scan_local(const float* __restrict__ dt, const float* __restrict__ xs,
                               const float* __restrict__ Bm, const float* __restrict__ Amat,
                               float* __restrict__ aggA, float* __restrict__ aggH) {
  extern __shared__ char smem[];
  float* dtc = (float*)smem;            // 128 x 20
  float* xsc = dtc + 128 * 20;
  float* bmc = xsc + 128 * 20;
  int tid = threadIdx.x;
  int chunk = blockIdx.x, dgrp = blockIdx.y, b = blockIdx.z;
  int t0 = b * Lc + chunk * CSc;
  int d0 = dgrp * 16;
  float4* dtc4 = (float4*)dtc; float4* xsc4 = (float4*)xsc; float4* bmc4 = (float4*)bmc;
  const float4* dtg = (const float4*)dt; const float4* xsg = (const float4*)xs;
  const float4* bmg = (const float4*)Bm;
  for (int i = tid; i < CSc * 4; i += 256) {
    int r = i >> 2, c = i & 3;
    dtc4[r * 5 + c] = dtg[(size_t)(t0 + r) * 24 + (d0 >> 2) + c];
    xsc4[r * 5 + c] = xsg[(size_t)(t0 + r) * 24 + (d0 >> 2) + c];
    bmc4[r * 5 + c] = bmg[(size_t)(t0 + r) * 4 + c];
  }
  __syncthreads();
  int s = tid & 15, dl = tid >> 4;
  float A = Amat[(d0 + dl) * DSc + s];
  float ap = 1.f, h = 0.f;
  for (int r = 0; r < CSc; ++r) {
    float dtv = dtc[r * 20 + dl];
    float a = __expf(dtv * A);
    h = a * h + dtv * xsc[r * 20 + dl] * bmc[r * 20 + s];
    ap *= a;
  }
  size_t idx = ((size_t)((b * DIc + d0 + dl) * DSc + s)) * NCHc + chunk;
  aggA[idx] = ap; aggH[idx] = h;
}

// ---------------- K3b: carry scan across chunks (3072 independent series) ----------------
__global__ void k3b_scan_carry(const float* __restrict__ aggA, const float* __restrict__ aggH,
                               float* __restrict__ carry) {
  int idx = blockIdx.x * blockDim.x + threadIdx.x;
  if (idx >= BB * DIc * DSc) return;
  const float* pa = aggA + (size_t)idx * NCHc;
  const float* ph = aggH + (size_t)idx * NCHc;
  float* pc = carry + (size_t)idx * NCHc;
  float c = 0.f;
  for (int ch = 0; ch < NCHc; ++ch) { pc[ch] = c; c = ph[ch] + pa[ch] * c; }
}

// ---------------- K3c: final scan pass, y = <h,C> + D*xs, gate with silu(z) ----------------
__global__ void k3c_scan_final(const float* __restrict__ dt, const float* __restrict__ xs,
                               const float* __restrict__ Bm, const float* __restrict__ Cm,
                               const float* __restrict__ z, const float* __restrict__ Dvec,
                               const float* __restrict__ Amat, const float* __restrict__ carry,
                               float* __restrict__ ygated) {
  extern __shared__ char smem[];
  float* dtc = (float*)smem;            // 4 x (128 x 20)
  float* xsc = dtc + 128 * 20;
  float* bmc = xsc + 128 * 20;
  float* cmc = bmc + 128 * 20;
  int tid = threadIdx.x;
  int chunk = blockIdx.x, dgrp = blockIdx.y, b = blockIdx.z;
  int t0 = b * Lc + chunk * CSc;
  int d0 = dgrp * 16;
  float4* dtc4 = (float4*)dtc; float4* xsc4 = (float4*)xsc;
  float4* bmc4 = (float4*)bmc; float4* cmc4 = (float4*)cmc;
  const float4* dtg = (const float4*)dt; const float4* xsg = (const float4*)xs;
  const float4* bmg = (const float4*)Bm; const float4* cmg = (const float4*)Cm;
  for (int i = tid; i < CSc * 4; i += 256) {
    int r = i >> 2, c = i & 3;
    dtc4[r * 5 + c] = dtg[(size_t)(t0 + r) * 24 + (d0 >> 2) + c];
    xsc4[r * 5 + c] = xsg[(size_t)(t0 + r) * 24 + (d0 >> 2) + c];
    bmc4[r * 5 + c] = bmg[(size_t)(t0 + r) * 4 + c];
    cmc4[r * 5 + c] = cmg[(size_t)(t0 + r) * 4 + c];
  }
  __syncthreads();
  int s = tid & 15, dl = tid >> 4;
  int d = d0 + dl;
  float A  = Amat[d * DSc + s];
  float h  = carry[((size_t)((b * DIc + d) * DSc + s)) * NCHc + chunk];
  float Dd = Dvec[d];
  for (int r = 0; r < CSc; ++r) {
    float dtv = dtc[r * 20 + dl];
    float a = __expf(dtv * A);
    h = a * h + dtv * xsc[r * 20 + dl] * bmc[r * 20 + s];
    float y = h * cmc[r * 20 + s];
    y += __shfl_xor(y, 1);
    y += __shfl_xor(y, 2);
    y += __shfl_xor(y, 4);
    y += __shfl_xor(y, 8);
    if (s == 0) {
      int t = t0 + r;
      float zv = z[(size_t)t * DIc + d];
      ygated[(size_t)t * DIc + d] = (y + Dd * xsc[r * 20 + dl]) * (zv * sigmoidf_(zv));
    }
  }
}

// ---------------- K4: out_proj (96 -> 64) + skip1 ----------------
__global__ void k4_outproj(const float* __restrict__ ygated, const __bf16* __restrict__ woutswz,
                           const float* __restrict__ x, const float* __restrict__ ss1,
                           float* __restrict__ h1) {
  extern __shared__ char smem[];
  __bf16* As = (__bf16*)smem;                 // 128 x 96 bf16 (frag-swizzled, KT=3)
  int tid = threadIdx.x;
  int t0  = blockIdx.x * 128;
  const float4* yg4 = (const float4*)ygated;
  for (int i = tid; i < 128 * 24; i += 256) {
    int r = i / 24, q = i - r * 24, d0 = q * 4;
    float4 v = yg4[(size_t)(t0 + r) * 24 + q];
    v4bf o;
    o[0] = (__bf16)v.x; o[1] = (__bf16)v.y; o[2] = (__bf16)v.z; o[3] = (__bf16)v.w;
    *(v4bf*)(As + aswz_idx(r, d0, 3)) = o;
  }
  __syncthreads();
  int wave = tid >> 5, lane = tid & 31;
  v8f acc[4] = {};
#pragma unroll
  for (int kt = 0; kt < 3; ++kt) {
    v16bf a = frag_ld(As + ((wave * 3 + kt) * 32 + lane) * 16);
#pragma unroll
    for (int nt = 0; nt < 4; ++nt)
      acc[nt] = wmma_bf16(a, frag_ld(woutswz + ((kt * 4 + nt) * 32 + lane) * 16), acc[nt]);
  }
#pragma unroll
  for (int nt = 0; nt < 4; ++nt) {
    int col = nt * 16 + (lane & 15);
    int rb  = t0 + wave * 16 + ((lane >> 4) << 3);
    float sc = ss1[col];
#pragma unroll
    for (int v = 0; v < 8; ++v) {
      int row = rb + v;
      h1[(size_t)row * Cc + col] = x[(size_t)row * Cc + col] * sc + acc[nt][v];
    }
  }
}

// ---------------- K5: LN2 + modulation + fc1 (64 -> 96) + exact gelu ----------------
__global__ void k5_ln_fc1(const float* __restrict__ h1, const float* __restrict__ w2e,
                          const float* __restrict__ b2e, const __bf16* __restrict__ wfc1swz,
                          const float* __restrict__ fc1b, float* __restrict__ a1) {
  extern __shared__ char smem[];
  float*  xsh = (float*)smem;                       // 128 x 68
  __bf16* As  = (__bf16*)(smem + 128 * 68 * 4);     // 128 x 64 (frag-swizzled, KT=2)
  float*  mu  = (float*)((char*)As + 128 * 64 * 2);
  float*  rs  = mu + 128;
  int tid = threadIdx.x;
  int t0  = blockIdx.x * 128;
  int bat = t0 / Lc;
  const float4* hg = (const float4*)h1;
  float4* xsh4 = (float4*)xsh;
  for (int i = tid; i < 128 * 16; i += 256) {
    int r = i >> 4, c = i & 15;
    xsh4[r * 17 + c] = hg[(size_t)(t0 + r) * 16 + c];
  }
  __syncthreads();
  if (tid < 128) {
    float s = 0.f, ss = 0.f;
    const float4* p = xsh4 + tid * 17;
#pragma unroll
    for (int c = 0; c < 16; ++c) {
      float4 v = p[c];
      s  += v.x + v.y + v.z + v.w;
      ss += v.x * v.x + v.y * v.y + v.z * v.z + v.w * v.w;
    }
    float m = s * (1.f / Cc);
    mu[tid] = m;
    rs[tid] = rsqrtf(ss * (1.f / Cc) - m * m + 1e-6f);
  }
  __syncthreads();
  for (int i = tid; i < 128 * 16; i += 256) {
    int r = i >> 4, c4 = i & 15, c0 = c4 * 4;
    float4 v = xsh4[r * 17 + c4];
    float m = mu[r], rr = rs[r];
    const float* we = w2e + bat * Cc + c0;
    const float* be = b2e + bat * Cc + c0;
    v4bf o;
    o[0] = (__bf16)((v.x - m) * rr * we[0] + be[0]);
    o[1] = (__bf16)((v.y - m) * rr * we[1] + be[1]);
    o[2] = (__bf16)((v.z - m) * rr * we[2] + be[2]);
    o[3] = (__bf16)((v.w - m) * rr * we[3] + be[3]);
    *(v4bf*)(As + aswz_idx(r, c0, 2)) = o;
  }
  __syncthreads();
  int wave = tid >> 5, lane = tid & 31;
  v8f acc[6] = {};
#pragma unroll
  for (int kt = 0; kt < 2; ++kt) {
    v16bf a = frag_ld(As + ((wave * 2 + kt) * 32 + lane) * 16);
#pragma unroll
    for (int nt = 0; nt < 6; ++nt)
      acc[nt] = wmma_bf16(a, frag_ld(wfc1swz + ((kt * 6 + nt) * 32 + lane) * 16), acc[nt]);
  }
#pragma unroll
  for (int nt = 0; nt < 6; ++nt) {
    int col = nt * 16 + (lane & 15);
    int rb  = t0 + wave * 16 + ((lane >> 4) << 3);
    float bia = fc1b[col];
#pragma unroll
    for (int v = 0; v < 8; ++v) {
      float g = acc[nt][v] + bia;
      a1[(size_t)(rb + v) * MHc + col] = 0.5f * g * (1.f + erff(g * 0.70710678f));
    }
  }
}

// ---------------- K6: fc2 (96 -> 64) + bias + skip2 -> out ----------------
__global__ void k6_fc2(const float* __restrict__ a1, const __bf16* __restrict__ wfc2swz,
                       const float* __restrict__ fc2b, const float* __restrict__ h1,
                       const float* __restrict__ ss2, float* __restrict__ out) {
  extern __shared__ char smem[];
  __bf16* As = (__bf16*)smem;                 // 128 x 96 (frag-swizzled, KT=3)
  int tid = threadIdx.x;
  int t0  = blockIdx.x * 128;
  const float4* ag = (const float4*)a1;
  for (int i = tid; i < 128 * 24; i += 256) {
    int r = i / 24, q = i - r * 24, d0 = q * 4;
    float4 v = ag[(size_t)(t0 + r) * 24 + q];
    v4bf o;
    o[0] = (__bf16)v.x; o[1] = (__bf16)v.y; o[2] = (__bf16)v.z; o[3] = (__bf16)v.w;
    *(v4bf*)(As + aswz_idx(r, d0, 3)) = o;
  }
  __syncthreads();
  int wave = tid >> 5, lane = tid & 31;
  v8f acc[4] = {};
#pragma unroll
  for (int kt = 0; kt < 3; ++kt) {
    v16bf a = frag_ld(As + ((wave * 3 + kt) * 32 + lane) * 16);
#pragma unroll
    for (int nt = 0; nt < 4; ++nt)
      acc[nt] = wmma_bf16(a, frag_ld(wfc2swz + ((kt * 4 + nt) * 32 + lane) * 16), acc[nt]);
  }
#pragma unroll
  for (int nt = 0; nt < 4; ++nt) {
    int col = nt * 16 + (lane & 15);
    int rb  = t0 + wave * 16 + ((lane >> 4) << 3);
    float bia = fc2b[col], sc = ss2[col];
#pragma unroll
    for (int v = 0; v < 8; ++v) {
      int row = rb + v;
      out[(size_t)row * Cc + col] = h1[(size_t)row * Cc + col] * sc + acc[nt][v] + bia;
    }
  }
}

// ---------------- host launcher ----------------
extern "C" void kernel_launch(void* const* d_in, const int* in_sizes, int n_in,
                              void* d_out, int out_size, void* d_ws, size_t ws_size,
                              hipStream_t stream) {
  (void)in_sizes; (void)n_in; (void)out_size; (void)ws_size;
  const float* x      = (const float*)d_in[0];
  const float* cdp    = (const float*)d_in[1];
  const float* vk1    = (const float*)d_in[2];
  const float* ln1w   = (const float*)d_in[3];
  const float* ln1b   = (const float*)d_in[4];
  const float* ss1    = (const float*)d_in[5];
  const float* inprj  = (const float*)d_in[6];
  const float* conv_w = (const float*)d_in[7];
  const float* conv_b = (const float*)d_in[8];
  const float* xprj   = (const float*)d_in[9];
  const float* dtw    = (const float*)d_in[10];
  const float* dtb    = (const float*)d_in[11];
  const float* A_log  = (const float*)d_in[12];
  const float* Dvec   = (const float*)d_in[13];
  const float* outprj = (const float*)d_in[14];
  const float* vk2    = (const float*)d_in[15];
  const float* ln2w   = (const float*)d_in[16];
  const float* ln2b   = (const float*)d_in[17];
  const float* ss2    = (const float*)d_in[18];
  const float* fc1w   = (const float*)d_in[19];
  const float* fc1bv  = (const float*)d_in[20];
  const float* fc2w   = (const float*)d_in[21];
  const float* fc2bv  = (const float*)d_in[22];

  float* wsf = (float*)d_ws;
  size_t o = 0;
  float* xs_pre = wsf + o; o += (size_t)NTc * DIc;
  float* zbuf   = wsf + o; o += (size_t)NTc * DIc;
  float* xsbuf  = wsf + o; o += (size_t)NTc * DIc;
  float* dtbuf  = wsf + o; o += (size_t)NTc * DIc;
  float* Bmb    = wsf + o; o += (size_t)NTc * DSc;
  float* Cmb    = wsf + o; o += (size_t)NTc * DSc;
  float* ygated = wsf + o; o += (size_t)NTc * DIc;
  float* h1buf  = wsf + o; o += (size_t)NTc * Cc;
  float* a1buf  = wsf + o; o += (size_t)NTc * MHc;
  float* aggA   = wsf + o; o += (size_t)BB * DIc * DSc * NCHc;
  float* aggH   = wsf + o; o += (size_t)BB * DIc * DSc * NCHc;
  float* carry  = wsf + o; o += (size_t)BB * DIc * DSc * NCHc;
  float* w1e = wsf + o; o += BB * Cc;
  float* b1e = wsf + o; o += BB * Cc;
  float* w2e = wsf + o; o += BB * Cc;
  float* b2e = wsf + o; o += BB * Cc;
  float* Amat = wsf + o; o += DIc * DSc;
  __bf16* bfb = (__bf16*)(wsf + o);
  size_t ob = 0;
  __bf16* inprojb  = bfb + ob; ob += Cc * N1c;
  __bf16* xprojb   = bfb + ob; ob += DIc * 48;
  __bf16* outprojb = bfb + ob; ob += DIc * Cc;
  __bf16* fc1bw    = bfb + ob; ob += Cc * MHc;
  __bf16* fc2bw    = bfb + ob; ob += MHc * Cc;

  const size_t SM1  = 128 * 68 * 4 + 128 * 64 * 2 + 256 * 4;              // 52224
  const size_t SM2  = 131 * 100 * 4 + 128 * 96 * 2 + 128 * 48 * 4;        // 101552
  const size_t SM3A = 3 * 128 * 20 * 4;                                   // 30720
  const size_t SM3C = 4 * 128 * 20 * 4;                                   // 40960
  const size_t SM4  = 128 * 96 * 2;                                       // 24576
  const size_t SM5  = 128 * 68 * 4 + 128 * 64 * 2 + 256 * 4;              // 52224
  const size_t SM6  = 128 * 96 * 2;                                       // 24576

  k0_prep<<<16, 256, 0, stream>>>(cdp, vk1, vk2, ln1w, ln1b, ln2w, ln2b, inprj, xprj, outprj,
                                  fc1w, fc2w, A_log, w1e, b1e, w2e, b2e, Amat,
                                  inprojb, xprojb, outprojb, fc1bw, fc2bw);
  k1_ln_inproj<<<NTc / 128, 256, SM1, stream>>>(x, w1e, b1e, inprojb, xs_pre, zbuf);
  k2_conv_xproj<<<NTc / 128, 256, SM2, stream>>>(xs_pre, conv_w, conv_b, xprojb, dtw, dtb,
                                                 xsbuf, dtbuf, Bmb, Cmb);
  k3a_scan_local<<<dim3(NCHc, DIc / 16, BB), 256, SM3A, stream>>>(dtbuf, xsbuf, Bmb, Amat, aggA, aggH);
  k3b_scan_carry<<<(BB * DIc * DSc + 255) / 256, 256, 0, stream>>>(aggA, aggH, carry);
  k3c_scan_final<<<dim3(NCHc, DIc / 16, BB), 256, SM3C, stream>>>(dtbuf, xsbuf, Bmb, Cmb, zbuf,
                                                                  Dvec, Amat, carry, ygated);
  k4_outproj<<<NTc / 128, 256, SM4, stream>>>(ygated, outprojb, x, ss1, h1buf);
  k5_ln_fc1<<<NTc / 128, 256, SM5, stream>>>(h1buf, w2e, b2e, fc1bw, fc1bv, a1buf);
  k6_fc2<<<NTc / 128, 256, SM6, stream>>>(a1buf, fc2bw, fc2bv, h1buf, ss2, (float*)d_out);
}
